// GAT_91259465105438
// MI455X (gfx1250) — matrix operand order repaired and checked by reading, not compiled, not run on hardware
//
#include <hip/hip_runtime.h>
#include <hip/hip_bf16.h>
#include <math.h>

// ---------------------------------------------------------------------------
// GAT (2x GATConv + BN + ELU, mean/max pool, 3-layer MLP head) for gfx1250.
// Dense feature GEMMs use V_WMMA_F32_16X16X32_F16 (f32 accumulate).
// Edge softmax/aggregation is the bandwidth bottleneck (~1GB irregular
// traffic/layer @ 23.3 TB/s HBM) and is done with float atomics.
// ---------------------------------------------------------------------------

typedef __attribute__((ext_vector_type(16))) _Float16 v16h;
typedef __attribute__((ext_vector_type(8)))  float    v8f;

#define HID 64
#define NHEAD 8
#define OUTC 8
#define NGRAPH 64
#define EPSBN 1e-5f

// Monotonic float<->uint encoding so atomicMax(uint) == float max (handles negatives).
__device__ __forceinline__ unsigned enc_f(float f) {
    unsigned u = __float_as_uint(f);
    return (u & 0x80000000u) ? ~u : (u | 0x80000000u);
}
__device__ __forceinline__ float dec_f(unsigned e) {
    return (e & 0x80000000u) ? __uint_as_float(e & 0x7FFFFFFFu) : __uint_as_float(~e);
}
#define ENC_NEG_INF 0x007FFFFFu   // enc(-inf)

// ------------------------------ fills --------------------------------------
__global__ void fill_f32(float* p, float v, int n) {
    int i = blockIdx.x * blockDim.x + threadIdx.x;
    if (i < n) p[i] = v;
}
__global__ void fill_u32(unsigned* p, unsigned v, int n) {
    int i = blockIdx.x * blockDim.x + threadIdx.x;
    if (i < n) p[i] = v;
}
__global__ void f32_to_f16(const float* __restrict__ s, _Float16* __restrict__ d, int n) {
    int i = blockIdx.x * blockDim.x + threadIdx.x;
    if (i < n) d[i] = (_Float16)s[i];
}

// ------------------------------ WMMA GEMM ----------------------------------
// Y[nrows,64] = X[nrows,KD] * Wh^T  (Wh is [64,KD] f16 row-major, i.e. B[k][n]=Wh[n][k])
// One wave handles a 16(M) x 64(N) strip; 4 waves / block.
template <int KD>
__global__ void __launch_bounds__(128)
gemm_wmma(const float* __restrict__ X, const _Float16* __restrict__ Wh,
          float* __restrict__ Y, int nrows) {
    const int wave = threadIdx.x >> 5;
    const int lane = threadIdx.x & 31;
    const int tile = blockIdx.x * 4 + wave;
    const int m0 = tile * 16;
    if (m0 >= nrows) return;                 // wave-uniform: EXEC all-ones inside

    const int r  = lane & 15;
    const int hi = lane >> 4;
    int arow = m0 + r; if (arow >= nrows) arow = nrows - 1;

    v8f acc0 = {}, acc1 = {}, acc2 = {}, acc3 = {};

#pragma unroll
    for (int kt = 0; kt < KD; kt += 32) {
        // A fragment: 16-bit A 16x32 layout. lane row = m0+r,
        // halves i -> K = kt + hi*16 + i (contiguous, matches B).
        const float* xp = X + (size_t)arow * KD + kt + hi * 16;
        float4 f0 = *(const float4*)(xp);
        float4 f1 = *(const float4*)(xp + 4);
        float4 f2 = *(const float4*)(xp + 8);
        float4 f3 = *(const float4*)(xp + 12);
        v16h av;
        av[0]  = (_Float16)f0.x; av[1]  = (_Float16)f0.y; av[2]  = (_Float16)f0.z; av[3]  = (_Float16)f0.w;
        av[4]  = (_Float16)f1.x; av[5]  = (_Float16)f1.y; av[6]  = (_Float16)f1.z; av[7]  = (_Float16)f1.w;
        av[8]  = (_Float16)f2.x; av[9]  = (_Float16)f2.y; av[10] = (_Float16)f2.z; av[11] = (_Float16)f2.w;
        av[12] = (_Float16)f3.x; av[13] = (_Float16)f3.y; av[14] = (_Float16)f3.z; av[15] = (_Float16)f3.w;

        // B fragments: lane column n = 16*t + r, halves i -> K = kt + hi*16 + i
        // => 16 contiguous f16 in Wh row (32B aligned).
        const _Float16* wb = Wh + kt + hi * 16;
        v16h b0 = *(const v16h*)(wb + (size_t)(r      ) * KD);
        v16h b1 = *(const v16h*)(wb + (size_t)(16 + r ) * KD);
        v16h b2 = *(const v16h*)(wb + (size_t)(32 + r ) * KD);
        v16h b3 = *(const v16h*)(wb + (size_t)(48 + r ) * KD);

        acc0 = __builtin_amdgcn_wmma_f32_16x16x32_f16(false, av, false, b0, (short)0, acc0, false, false);
        acc1 = __builtin_amdgcn_wmma_f32_16x16x32_f16(false, av, false, b1, (short)0, acc1, false, false);
        acc2 = __builtin_amdgcn_wmma_f32_16x16x32_f16(false, av, false, b2, (short)0, acc2, false, false);
        acc3 = __builtin_amdgcn_wmma_f32_16x16x32_f16(false, av, false, b3, (short)0, acc3, false, false);
    }

    // C/D layout: VGPR j -> M = j + 8*hi, N = r (within each 16-col tile).
#pragma unroll
    for (int j = 0; j < 8; j++) {
        int row = m0 + hi * 8 + j;
        if (row < nrows) {
            float* yp = Y + (size_t)row * HID + r;
            yp[0]  = acc0[j];
            yp[16] = acc1[j];
            yp[32] = acc2[j];
            yp[48] = acc3[j];
        }
    }
}

// ---------------------- per-node attention scalars -------------------------
// AS[n,h] = sum_o H[n,h,o]*att_s[h,o] ; AD likewise.
__global__ void attn_scores(const float* __restrict__ H, const float* __restrict__ atts,
                            const float* __restrict__ attd, float* __restrict__ AS,
                            float* __restrict__ AD, int nN) {
    int idx = blockIdx.x * blockDim.x + threadIdx.x;
    if (idx >= nN * NHEAD) return;
    int n = idx >> 3, h = idx & 7;
    const float* hp = H + (size_t)n * HID + h * OUTC;
    float s = 0.f, d = 0.f;
#pragma unroll
    for (int o = 0; o < OUTC; o++) {
        s += hp[o] * atts[h * OUTC + o];
        d += hp[o] * attd[h * OUTC + o];
    }
    AS[idx] = s; AD[idx] = d;
}

// ------------------------- edge pass 1: segment max ------------------------
// one thread per (edge, head); edges >= E are the self-loops.
__global__ void edge_max(const int* __restrict__ src, const int* __restrict__ dst,
                         int E, int nN, const float* __restrict__ AS,
                         const float* __restrict__ AD, unsigned* __restrict__ EMAX) {
    int idx = blockIdx.x * blockDim.x + threadIdx.x;
    if (idx >= (E + nN) * NHEAD) return;
    int e = idx >> 3, h = idx & 7;
    int s, d;
    if (e < E) { s = src[e]; d = dst[e]; } else { s = e - E; d = s; }
    float v = AS[s * NHEAD + h] + AD[d * NHEAD + h];
    v = v > 0.f ? v : 0.2f * v;                       // leaky_relu(0.2)
    atomicMax(EMAX + d * NHEAD + h, enc_f(v));
}

// ---------------- edge pass 2: exp + denom + weighted scatter --------------
__global__ void edge_sum(const int* __restrict__ src, const int* __restrict__ dst,
                         int E, int nN, const float* __restrict__ AS,
                         const float* __restrict__ AD, const unsigned* __restrict__ EMAX,
                         float* __restrict__ DEN, const float* __restrict__ H,
                         float* __restrict__ AGG) {
    int idx = blockIdx.x * blockDim.x + threadIdx.x;
    if (idx >= (E + nN) * NHEAD) return;
    int e = idx >> 3, h = idx & 7;
    int s, d;
    if (e < E) { s = src[e]; d = dst[e]; } else { s = e - E; d = s; }
    float v = AS[s * NHEAD + h] + AD[d * NHEAD + h];
    v = v > 0.f ? v : 0.2f * v;
    float ex = __expf(v - dec_f(EMAX[d * NHEAD + h]));
    atomicAdd(DEN + d * NHEAD + h, ex);
    const float* hs = H + (size_t)s * HID + h * OUTC;
    float* ag = AGG + (size_t)d * HID + h * OUTC;
#pragma unroll
    for (int o = 0; o < OUTC; o++) atomicAdd(ag + o, ex * hs[o]);
}

// ------------------ node finalize: /denom + bias + BN + ELU ----------------
__global__ void finalize_node(const float* __restrict__ AGG, const float* __restrict__ DEN,
                              const float* __restrict__ bias, const float* __restrict__ g,
                              const float* __restrict__ b, const float* __restrict__ rm,
                              const float* __restrict__ rv, float* __restrict__ OUT, int nN) {
    int idx = blockIdx.x * blockDim.x + threadIdx.x;
    if (idx >= nN * HID) return;
    int n = idx >> 6, c = idx & 63, h = c >> 3;
    float v = AGG[idx] / DEN[n * NHEAD + h] + bias[c];
    v = (v - rm[c]) * rsqrtf(rv[c] + EPSBN) * g[c] + b[c];
    OUT[idx] = v > 0.f ? v : expm1f(v);               // ELU
}

// ------------------------------- pooling -----------------------------------
__global__ void pool_kernel(const float* __restrict__ HP, const int* __restrict__ batch,
                            float* __restrict__ PSUM, unsigned* __restrict__ PMAX,
                            float* __restrict__ PCNT, int nN) {
    int idx = blockIdx.x * blockDim.x + threadIdx.x;
    if (idx >= nN * HID) return;
    int n = idx >> 6, c = idx & 63;
    int gid = batch[n];
    float v = HP[idx];
    atomicAdd(PSUM + gid * HID + c, v);
    atomicMax(PMAX + gid * HID + c, enc_f(v));
    if (c == 0) atomicAdd(PCNT + gid, 1.0f);
}

// ------------------------------- MLP head ----------------------------------
// B=64 rows; one thread per graph. Trivial FLOPs; weights stay L0/L2-hot.
__global__ void mlp_head(const float* __restrict__ PSUM, const unsigned* __restrict__ PMAX,
                         const float* __restrict__ PCNT,
                         const float* fc1w, const float* fc1b,
                         const float* g1, const float* b1, const float* m1, const float* v1,
                         const float* fc2w, const float* fc2b,
                         const float* g2, const float* b2, const float* m2, const float* v2,
                         const float* fc3w, const float* fc3b, float* __restrict__ out) {
    int t = threadIdx.x;
    if (t >= NGRAPH) return;
    float cnt = fmaxf(PCNT[t], 1.0f);

    float h1v[64];
    for (int j = 0; j < 64; j++) {
        float a = fc1b[j];
        const float* wr = fc1w + j * 128;
        for (int k = 0; k < 64; k++) a += wr[k] * (PSUM[t * HID + k] / cnt);
        for (int k = 0; k < 64; k++) a += wr[64 + k] * dec_f(PMAX[t * HID + k]);
        a = (a - m1[j]) * rsqrtf(v1[j] + EPSBN) * g1[j] + b1[j];
        h1v[j] = fmaxf(a, 0.f);
    }
    float h2v[32];
    for (int j = 0; j < 32; j++) {
        float a = fc2b[j];
        const float* wr = fc2w + j * 64;
        for (int k = 0; k < 64; k++) a += wr[k] * h1v[k];
        a = (a - m2[j]) * rsqrtf(v2[j] + EPSBN) * g2[j] + b2[j];
        h2v[j] = fmaxf(a, 0.f);
    }
    for (int j = 0; j < 2; j++) {
        float a = fc3b[j];
        const float* wr = fc3w + j * 32;
        for (int k = 0; k < 32; k++) a += wr[k] * h2v[k];
        out[t * 2 + j] = a;
    }
}

// ---------------------------------------------------------------------------
extern "C" void kernel_launch(void* const* d_in, const int* in_sizes, int n_in,
                              void* d_out, int out_size, void* d_ws, size_t ws_size,
                              hipStream_t stream) {
    const int N = in_sizes[0] / 128;      // 50000
    const int E = in_sizes[1] / 2;        // 1.6M

    const float* x    = (const float*)d_in[0];
    const int*   eidx = (const int*)d_in[1];
    const int*   batch= (const int*)d_in[2];
    const float* W1   = (const float*)d_in[3];
    const float* as1  = (const float*)d_in[4];
    const float* ad1  = (const float*)d_in[5];
    const float* b1   = (const float*)d_in[6];
    const float* W2   = (const float*)d_in[7];
    const float* as2  = (const float*)d_in[8];
    const float* ad2  = (const float*)d_in[9];
    const float* b2   = (const float*)d_in[10];
    const float* bn1g = (const float*)d_in[11];
    const float* bn1b = (const float*)d_in[12];
    const float* bn1m = (const float*)d_in[13];
    const float* bn1v = (const float*)d_in[14];
    const float* bn2g = (const float*)d_in[15];
    const float* bn2b = (const float*)d_in[16];
    const float* bn2m = (const float*)d_in[17];
    const float* bn2v = (const float*)d_in[18];
    const float* fc1w = (const float*)d_in[19];
    const float* fc1b = (const float*)d_in[20];
    const float* bf1g = (const float*)d_in[21];
    const float* bf1b = (const float*)d_in[22];
    const float* bf1m = (const float*)d_in[23];
    const float* bf1v = (const float*)d_in[24];
    const float* fc2w = (const float*)d_in[25];
    const float* fc2b = (const float*)d_in[26];
    const float* bf2g = (const float*)d_in[27];
    const float* bf2b = (const float*)d_in[28];
    const float* bf2m = (const float*)d_in[29];
    const float* bf2v = (const float*)d_in[30];
    const float* fc3w = (const float*)d_in[31];
    const float* fc3b = (const float*)d_in[32];

    const int* src = eidx;
    const int* dst = eidx + E;

    // ---- workspace carve-up (d_ws is 256B aligned from hipMalloc) ----
    float*    H    = (float*)d_ws;                        // [N,64] linear features
    float*    HP   = H    + (size_t)N * HID;              // [N,64] post BN+ELU
    float*    AGG  = HP   + (size_t)N * HID;              // [N,64] scatter accumulator
    float*    AS   = AGG  + (size_t)N * HID;              // [N,8]
    float*    AD   = AS   + (size_t)N * NHEAD;            // [N,8]
    unsigned* EMAX = (unsigned*)(AD + (size_t)N * NHEAD); // [N,8] encoded max
    float*    DEN  = (float*)(EMAX + (size_t)N * NHEAD);  // [N,8]
    float*    PSUM = DEN  + (size_t)N * NHEAD;            // [64,64]
    unsigned* PMAX = (unsigned*)(PSUM + NGRAPH * HID);    // [64,64]
    float*    PCNT = (float*)(PMAX + NGRAPH * HID);       // [64]
    _Float16* W1h  = (_Float16*)(PCNT + NGRAPH);          // [64,128] f16
    _Float16* W2h  = W1h + HID * 128;                     // [64,64] f16

    const int TB = 256;
    const int nodeHidT = N * HID, nodeHeadT = N * NHEAD;
    const int edgeT = (E + N) * NHEAD;
    const int gemmBlocks = ((N + 15) / 16 + 3) / 4;

    // weight conversion to f16 (once per call; deterministic)
    f32_to_f16<<<(HID * 128 + TB - 1) / TB, TB, 0, stream>>>(W1, W1h, HID * 128);
    f32_to_f16<<<(HID * 64 + TB - 1) / TB, TB, 0, stream>>>(W2, W2h, HID * 64);

    // =================== layer 1 ===================
    gemm_wmma<128><<<gemmBlocks, 128, 0, stream>>>(x, W1h, H, N);
    attn_scores<<<(nodeHeadT + TB - 1) / TB, TB, 0, stream>>>(H, as1, ad1, AS, AD, N);
    fill_u32<<<(nodeHeadT + TB - 1) / TB, TB, 0, stream>>>(EMAX, ENC_NEG_INF, nodeHeadT);
    fill_f32<<<(nodeHeadT + TB - 1) / TB, TB, 0, stream>>>(DEN, 0.f, nodeHeadT);
    fill_f32<<<(nodeHidT + TB - 1) / TB, TB, 0, stream>>>(AGG, 0.f, nodeHidT);
    edge_max<<<(edgeT + TB - 1) / TB, TB, 0, stream>>>(src, dst, E, N, AS, AD, EMAX);
    edge_sum<<<(edgeT + TB - 1) / TB, TB, 0, stream>>>(src, dst, E, N, AS, AD, EMAX, DEN, H, AGG);
    finalize_node<<<(nodeHidT + TB - 1) / TB, TB, 0, stream>>>(AGG, DEN, b1, bn1g, bn1b, bn1m, bn1v, HP, N);

    // =================== layer 2 ===================
    gemm_wmma<64><<<gemmBlocks, 128, 0, stream>>>(HP, W2h, H, N);
    attn_scores<<<(nodeHeadT + TB - 1) / TB, TB, 0, stream>>>(H, as2, ad2, AS, AD, N);
    fill_u32<<<(nodeHeadT + TB - 1) / TB, TB, 0, stream>>>(EMAX, ENC_NEG_INF, nodeHeadT);
    fill_f32<<<(nodeHeadT + TB - 1) / TB, TB, 0, stream>>>(DEN, 0.f, nodeHeadT);
    fill_f32<<<(nodeHidT + TB - 1) / TB, TB, 0, stream>>>(AGG, 0.f, nodeHidT);
    edge_max<<<(edgeT + TB - 1) / TB, TB, 0, stream>>>(src, dst, E, N, AS, AD, EMAX);
    edge_sum<<<(edgeT + TB - 1) / TB, TB, 0, stream>>>(src, dst, E, N, AS, AD, EMAX, DEN, H, AGG);
    finalize_node<<<(nodeHidT + TB - 1) / TB, TB, 0, stream>>>(AGG, DEN, b2, bn2g, bn2b, bn2m, bn2v, HP, N);

    // =================== pooling + head ===================
    fill_f32<<<(NGRAPH * HID + TB - 1) / TB, TB, 0, stream>>>(PSUM, 0.f, NGRAPH * HID);
    fill_u32<<<(NGRAPH * HID + TB - 1) / TB, TB, 0, stream>>>(PMAX, ENC_NEG_INF, NGRAPH * HID);
    fill_f32<<<(NGRAPH + TB - 1) / TB, TB, 0, stream>>>(PCNT, 0.f, NGRAPH);
    pool_kernel<<<(nodeHidT + TB - 1) / TB, TB, 0, stream>>>(HP, batch, PSUM, PMAX, PCNT, N);

    mlp_head<<<1, 64, 0, stream>>>(PSUM, PMAX, PCNT,
                                   fc1w, fc1b, bf1g, bf1b, bf1m, bf1v,
                                   fc2w, fc2b, bf2g, bf2b, bf2m, bf2v,
                                   fc3w, fc3b, (float*)d_out);
}
